// ChaoticGRU_65240553226437
// MI455X (gfx1250) — matrix operand order repaired and checked by reading, not compiled, not run on hardware
//
#include <hip/hip_runtime.h>
#include <hip/hip_bf16.h>

// GRU: BS=64, SEQ=512, IN=512, HID=1024.
//   Phase 1 (parallel): giNi[B*T, 3072] = x[B*T,512] @ [Wi | Wni] + bias  (bf16 WMMA, fp32 acc,
//            even/odd double-buffered fragment prefetch -> no register-rotation movs).
//   Phase 2 (scan): persistent kernel, 64 WGs x 4 waves. WG j owns hidden cols j*16..j*16+15.
//            Its 96KB recurrent-weight slice is staged into LDS ONCE (fragment-ordered layout);
//            every timestep: epilogue operands prefetched first, then 32 k-steps of
//            3 WMMAs each (B from LDS, A double-buffered from L2), elementwise GRU update
//            in registers, one device-wide atomic barrier per step.

typedef __attribute__((ext_vector_type(16))) __bf16 v16bf;
typedef __attribute__((ext_vector_type(8)))  float  v8f;

#define HID  1024
#define SEQ  512
#define BS   64
#define INP  512
#define N3   3072          // 2*HID + HID
#define N2   2048
#define KITER (HID / 32)   // 32 k-steps in the scan
#define FRAG_HALFS 512     // one 32x16 B fragment = 32 lanes * 16 halfs

static __device__ __forceinline__ int lane_id() { return threadIdx.x & 31; }

// A fragment: 16x32 bf16 tile from row-major memory (leading dim ld).
// lane<16 -> row=lane, K=0..15 contiguous; lane>=16 -> same row, K=16..31.
static __device__ __forceinline__ v16bf load_frag_a(const __bf16* p, int ld) {
  int l   = lane_id();
  int row = l & 15;
  int kb  = (l >> 4) << 4;            // 0 or 16
  const __bf16* c0 = p + (size_t)row * ld + kb;
  const __bf16* c1 = c0 + 8;
  v16bf f;
#pragma unroll
  for (int i = 0; i < 8; ++i) { f[i] = c0[i]; f[8 + i] = c1[i]; }
  return f;
}

// B fragment: 32x16 bf16 tile (KxN) from pre-transposed [N,K] row-major weights.
// lane<16 -> col=lane, K=0..15 contiguous; lane>=16 -> col, K=16..31.
static __device__ __forceinline__ v16bf load_frag_b(const __bf16* p, int ld) {
  int l   = lane_id();
  int col = l & 15;
  int kb  = (l >> 4) << 4;            // 0 or 16
  const __bf16* c = p + (size_t)col * ld + kb;
  v16bf f;
#pragma unroll
  for (int i = 0; i < 16; ++i) f[i] = c[i];
  return f;
}

// B fragment from LDS, fragment-ordered layout: frag base + lane*16 halfs (32B/lane).
static __device__ __forceinline__ v16bf load_frag_b_lds(const __bf16* p) {
  v16bf f;
#pragma unroll
  for (int i = 0; i < 16; ++i) f[i] = p[i];
  return f;
}

static __device__ __forceinline__ v8f wmma_bf16(v16bf a, v16bf b, v8f c) {
  return __builtin_amdgcn_wmma_f32_16x16x32_bf16(false, a, false, b, (short)0, c, false, false);
}

static __device__ __forceinline__ float sigmoidf_(float x) {
  return 1.0f / (1.0f + __expf(-x));
}

// ---------------- conversion / transpose kernels ----------------

__global__ void cvt_f32_to_bf16(const float* __restrict__ in, __bf16* __restrict__ out, int n) {
  for (int i = blockIdx.x * blockDim.x + threadIdx.x; i < n; i += gridDim.x * blockDim.x)
    out[i] = (__bf16)in[i];
}

// WinT[n][k] = (n<2048 ? Wi[k][n] : Wni[k][n-2048]),  shape [3072, 512] bf16
__global__ void build_WinT(const float* __restrict__ Wi, const float* __restrict__ Wni,
                           __bf16* __restrict__ WinT) {
  int total = N3 * INP;
  for (int i = blockIdx.x * blockDim.x + threadIdx.x; i < total; i += gridDim.x * blockDim.x) {
    int n = i / INP, k = i % INP;
    float v = (n < N2) ? Wi[(size_t)k * N2 + n] : Wni[(size_t)k * HID + (n - N2)];
    WinT[i] = (__bf16)v;
  }
}

// WrecT[n][k] = (n<2048 ? Wh[k][n] : Wnh[k][n-2048]),  shape [3072, 1024] bf16
__global__ void build_WrecT(const float* __restrict__ Wh, const float* __restrict__ Wnh,
                            __bf16* __restrict__ WrecT) {
  int total = N3 * HID;
  for (int i = blockIdx.x * blockDim.x + threadIdx.x; i < total; i += gridDim.x * blockDim.x) {
    int n = i / HID, k = i % HID;
    float v = (n < N2) ? Wh[(size_t)k * N2 + n] : Wnh[(size_t)k * HID + (n - N2)];
    WrecT[i] = (__bf16)v;
  }
}

// ---------------- phase 1: giNi = x @ [Wi|Wni] + bias ----------------
// M=32768, N=3072, K=512. One wave -> 16x64 output block (4 WMMA tiles, A reused 4x).
// Even/odd double buffering: stage 0 computes on {a0,b0} while {a1,b1} load, and vice
// versa -> fixed register assignment, no rotation movs. K/32 = 16 stages (even count).

__global__ void precompute_gemm(const __bf16* __restrict__ xbf,   // [BS*SEQ, INP]
                                const __bf16* __restrict__ WinT,  // [N3, INP]
                                const float*  __restrict__ Bg,    // [2048]
                                const float*  __restrict__ Bni,   // [1024]
                                float* __restrict__ giNi) {       // [BS*SEQ, N3]
  int wave = threadIdx.x >> 5;
  int task = blockIdx.x * 8 + wave;
  const int NBLK = N3 / 64;            // 48
  int mt = task / NBLK;
  int n0 = (task % NBLK) * 64;

  const __bf16* abase = xbf + (size_t)mt * 16 * INP;
  const __bf16* bbase = WinT + (size_t)n0 * INP;

  v8f acc[4] = {{}, {}, {}, {}};

  v16bf a0 = load_frag_a(abase, INP);
  v16bf b0[4];
#pragma unroll
  for (int s = 0; s < 4; ++s) b0[s] = load_frag_b(bbase + (size_t)s * 16 * INP, INP);

  for (int k = 0; k < INP; k += 64) {
    // prefetch odd stage (k+32 always < INP: 16 stages, even count)
    v16bf a1 = load_frag_a(abase + k + 32, INP);
    v16bf b1[4];
#pragma unroll
    for (int s = 0; s < 4; ++s) b1[s] = load_frag_b(bbase + (size_t)s * 16 * INP + k + 32, INP);

#pragma unroll
    for (int s = 0; s < 4; ++s) acc[s] = wmma_bf16(a0, b0[s], acc[s]);

    if (k + 64 < INP) {                // prefetch next even stage
      a0 = load_frag_a(abase + k + 64, INP);
#pragma unroll
      for (int s = 0; s < 4; ++s) b0[s] = load_frag_b(bbase + (size_t)s * 16 * INP + k + 64, INP);
    }

#pragma unroll
    for (int s = 0; s < 4; ++s) acc[s] = wmma_bf16(a1, b1[s], acc[s]);
  }

  int l = lane_id();
  int col = l & 15;
  int rb  = (l >> 4) << 3;             // C/D layout: VGPR v -> row v (lanes<16) / v+8 (lanes>=16)
#pragma unroll
  for (int s = 0; s < 4; ++s) {
    int gc = n0 + s * 16 + col;
    float bias = (gc < N2) ? Bg[gc] : Bni[gc - N2];
#pragma unroll
    for (int v = 0; v < 8; ++v) {
      int gm = mt * 16 + rb + v;
      giNi[(size_t)gm * N3 + gc] = acc[s][v] + bias;
    }
  }
}

// ---------------- phase 2: persistent GRU scan ----------------
// grid = 64 blocks (one per hidden-column tile), block = 128 threads (4 waves = 4 m-tiles).
// Dynamic LDS: 3 tiles * 32 k-iters * 1024B fragments = 96KB (fits CDNA5's 320KB/WG).

__global__ void gru_scan(const __bf16* __restrict__ WrecT, // [N3, HID] (rows: r | z | n)
                         const float*  __restrict__ giNi,  // [BS*SEQ, N3]
                         const float*  __restrict__ Bnh,   // [1024]
                         float*  __restrict__ hstate,      // [BS, HID] fp32
                         __bf16* __restrict__ hbf0,        // [BS, HID] ping
                         __bf16* __restrict__ hbf1,        // [BS, HID] pong
                         float*  __restrict__ out,         // [BS, SEQ, HID]
                         float*  __restrict__ htout,       // [BS, HID]
                         unsigned int* __restrict__ ctr,   // [SEQ] barrier counters (pre-zeroed)
                         int nblocks) {
  extern __shared__ __bf16 bsh[];      // [3][KITER][FRAG_HALFS]

  int j    = blockIdx.x;               // hidden column tile 0..63
  int wave = threadIdx.x >> 5;         // batch m-tile 0..3
  int l    = lane_id();
  int col  = l & 15;
  int jc   = j * 16 + col;
  int rb   = (l >> 4) << 3;
  float bnh = Bnh[jc];

  // ---- stage this WG's recurrent-weight slice to LDS once (fragment-ordered) ----
  // wave tau in {0,1,2} handles the tau-th gate block (r / z / n).
  if (wave < 3) {
    const __bf16* src = WrecT + ((size_t)wave * HID + j * 16 + col) * HID + ((l >> 4) << 4);
    __bf16* dst = bsh + (size_t)wave * KITER * FRAG_HALFS + (size_t)l * 16;
    for (int kk = 0; kk < KITER; ++kk) {
      const __bf16* s = src + kk * 32;
      __bf16* d = dst + (size_t)kk * FRAG_HALFS;
#pragma unroll
      for (int i = 0; i < 16; ++i) d[i] = s[i];
    }
  }
  __syncthreads();

  const __bf16* Br = bsh + 0 * KITER * FRAG_HALFS + (size_t)l * 16;
  const __bf16* Bz = bsh + 1 * KITER * FRAG_HALFS + (size_t)l * 16;
  const __bf16* Bn = bsh + 2 * KITER * FRAG_HALFS + (size_t)l * 16;

  for (int t = 0; t < SEQ; ++t) {
    const __bf16* hin  = (t & 1) ? hbf1 : hbf0;
    __bf16*       hout = (t & 1) ? hbf0 : hbf1;

    // ---- issue epilogue operand loads early: latency overlaps the 96 WMMAs below ----
    float gR[8], gZ[8], gN[8], hp[8];
#pragma unroll
    for (int v = 0; v < 8; ++v) {
      int b = wave * 16 + rb + v;
      size_t gbase = (size_t)(b * SEQ + t) * N3;
      gR[v] = giNi[gbase + jc];
      gZ[v] = giNi[gbase + HID + jc];
      gN[v] = giNi[gbase + 2 * HID + jc];
      hp[v] = hstate[(size_t)b * HID + jc];
    }

    v8f aR = {}, aZ = {}, aN = {};
    const __bf16* abase = hin + (size_t)wave * 16 * HID;

    // even/odd double-buffered A stream (KITER = 32, even)
    v16bf a0 = load_frag_a(abase, HID);
    for (int kk = 0; kk < KITER; kk += 2) {
      v16bf a1 = load_frag_a(abase + (kk + 1) * 32, HID);
      aR = wmma_bf16(a0, load_frag_b_lds(Br + (size_t)kk * FRAG_HALFS), aR);
      aZ = wmma_bf16(a0, load_frag_b_lds(Bz + (size_t)kk * FRAG_HALFS), aZ);
      aN = wmma_bf16(a0, load_frag_b_lds(Bn + (size_t)kk * FRAG_HALFS), aN);
      if (kk + 2 < KITER) a0 = load_frag_a(abase + (kk + 2) * 32, HID);
      aR = wmma_bf16(a1, load_frag_b_lds(Br + (size_t)(kk + 1) * FRAG_HALFS), aR);
      aZ = wmma_bf16(a1, load_frag_b_lds(Bz + (size_t)(kk + 1) * FRAG_HALFS), aZ);
      aN = wmma_bf16(a1, load_frag_b_lds(Bn + (size_t)(kk + 1) * FRAG_HALFS), aN);
    }

#pragma unroll
    for (int v = 0; v < 8; ++v) {
      int b = wave * 16 + rb + v;                           // batch row
      float r  = sigmoidf_(aR[v] + gR[v]);
      float z  = sigmoidf_(aZ[v] + gZ[v]);
      float n  = tanhf(gN[v] + r * (aN[v] + bnh));
      float hn = (1.0f - z) * n + z * hp[v];
      hstate[(size_t)b * HID + jc] = hn;
      hout[(size_t)b * HID + jc]   = (__bf16)hn;
      out[((size_t)b * SEQ + t) * HID + jc] = hn;
      if (t == SEQ - 1) htout[(size_t)b * HID + jc] = hn;
    }

    // device-wide barrier between timesteps (per-step counter, no reset needed)
    __syncthreads();
    __threadfence();
    if (threadIdx.x == 0) {
      __hip_atomic_fetch_add(&ctr[t], 1u, __ATOMIC_RELAXED, __HIP_MEMORY_SCOPE_AGENT);
      while (__hip_atomic_load(&ctr[t], __ATOMIC_RELAXED, __HIP_MEMORY_SCOPE_AGENT) <
             (unsigned)nblocks)
        __builtin_amdgcn_s_sleep(8);
    }
    __syncthreads();
    __threadfence();
  }
}

// ---------------- host launch ----------------

extern "C" void kernel_launch(void* const* d_in, const int* in_sizes, int n_in,
                              void* d_out, int out_size, void* d_ws, size_t ws_size,
                              hipStream_t stream) {
  const float* x   = (const float*)d_in[0];   // [64,512,512]
  const float* Wi  = (const float*)d_in[1];   // [512,2048]
  const float* Wh  = (const float*)d_in[2];   // [1024,2048]
  const float* Bg  = (const float*)d_in[3];   // [2048]
  const float* Wni = (const float*)d_in[4];   // [512,1024]
  const float* Wnh = (const float*)d_in[5];   // [1024,1024]
  const float* Bni = (const float*)d_in[6];   // [1024]
  const float* Bnh = (const float*)d_in[7];   // [1024]

  float* out   = (float*)d_out;                       // [64,512,1024]
  float* htout = out + (size_t)BS * SEQ * HID;        // [64,1024]

  // workspace carve-up (~446 MB)
  char* ws = (char*)d_ws;
  size_t off = 0;
  auto carve = [&](size_t bytes) -> char* {
    char* p = ws + off;
    off = (off + bytes + 255) & ~(size_t)255;
    return p;
  };
  const size_t nX = (size_t)BS * SEQ * INP;           // 16.78M
  __bf16* xbf   = (__bf16*)carve(nX * 2);
  __bf16* WinT  = (__bf16*)carve((size_t)N3 * INP * 2);
  __bf16* WrecT = (__bf16*)carve((size_t)N3 * HID * 2);
  float*  giNi  = (float*) carve((size_t)BS * SEQ * N3 * 4);   // 403 MB
  float*  hst   = (float*) carve((size_t)BS * HID * 4);
  __bf16* hbf0  = (__bf16*)carve((size_t)BS * HID * 2);
  __bf16* hbf1  = (__bf16*)carve((size_t)BS * HID * 2);
  unsigned int* ctr = (unsigned int*)carve(SEQ * sizeof(unsigned int));

  // zero h state, ping-pong bf16 state, and per-step barrier counters (capture-safe)
  hipMemsetAsync(hst,  0, (size_t)BS * HID * 4, stream);
  hipMemsetAsync(hbf0, 0, (size_t)BS * HID * 2, stream);
  hipMemsetAsync(hbf1, 0, (size_t)BS * HID * 2, stream);
  hipMemsetAsync(ctr,  0, SEQ * sizeof(unsigned int), stream);

  // conversions / weight transposes
  cvt_f32_to_bf16<<<4096, 256, 0, stream>>>(x, xbf, (int)nX);
  build_WinT<<<2048, 256, 0, stream>>>(Wi, Wni, WinT);
  build_WrecT<<<4096, 256, 0, stream>>>(Wh, Wnh, WrecT);

  // phase 1: fused input-projection GEMM (WMMA, even/odd pipelined)
  precompute_gemm<<<12288, 256, 0, stream>>>(xbf, WinT, Bg, Bni, giNi);

  // phase 2: persistent recurrent scan (WMMA, B-weights resident in 96KB LDS)
  const size_t ldsBytes = (size_t)3 * KITER * FRAG_HALFS * sizeof(__bf16);  // 96KB
  gru_scan<<<64, 128, ldsBytes, stream>>>(WrecT, giNi, Bnh, hst, hbf0, hbf1, out, htout, ctr, 64);

  (void)in_sizes; (void)n_in; (void)out_size; (void)ws_size;
}
